// SpikingClassifier_30949534335148
// MI455X (gfx1250) — matrix operand order repaired and checked by reading, not compiled, not run on hardware
//
#include <hip/hip_runtime.h>

typedef __attribute__((ext_vector_type(16))) _Float16 v16h;
typedef __attribute__((ext_vector_type(8)))  _Float16 h8;
typedef __attribute__((ext_vector_type(4)))  _Float16 h4;
typedef __attribute__((ext_vector_type(8)))  float    v8f;
typedef __attribute__((ext_vector_type(4)))  float    f4;

#define T_STEPS 14
#define B_TOT   4096
#define I_DIM   2048
#define H_DIM   1024
#define O_DIM   128
#define B_TILE  16
#define WAVES   16         // == B_TILE: each wave stages one batch row
#define NT      4          // n-tiles per wave: 64 tiles / 16 waves

#define WIN_HALVES  (H_DIM * I_DIM)     // w_in  f16
#define WREC_HALVES (H_DIM * H_DIM)     // w_rec f16
#define OW_HALVES   (O_DIM * H_DIM)     // out_w f16

// ---------------- weight fp32 -> f16 pre-conversion ----------------
__global__ void convert_weights(const float* __restrict__ w_in,
                                const float* __restrict__ w_rec,
                                const float* __restrict__ out_w,
                                _Float16* __restrict__ ws) {
    size_t idx = (size_t)blockIdx.x * 256 + threadIdx.x;
    const size_t total = (size_t)WIN_HALVES + WREC_HALVES + OW_HALVES;
    if (idx >= total) return;
    float v;
    if (idx < WIN_HALVES)                      v = w_in[idx];
    else if (idx < WIN_HALVES + WREC_HALVES)   v = w_rec[idx - WIN_HALVES];
    else                                       v = out_w[idx - WIN_HALVES - WREC_HALVES];
    ws[idx] = (_Float16)v;
}

// A fragment (16x32 f16) from f16 LDS; p = row_base + k0 + hi*16
// halves 0..15 <- p[0..15] (K = k0+hi*16+..)
__device__ __forceinline__ v16h load_a_h(const _Float16* p) {
    h8 lo = *(const h8*)(p);
    h8 hv = *(const h8*)(p + 8);
    return __builtin_shufflevector(lo, hv, 0,1,2,3,4,5,6,7,8,9,10,11,12,13,14,15);
}

__launch_bounds__(WAVES * 32)
__global__ void lif_kernel(const float* __restrict__ x,
                           const float* __restrict__ out_b,
                           const _Float16* __restrict__ ws,
                           float* __restrict__ out) {
    __shared__ __align__(16) _Float16 xs[B_TILE * I_DIM];  // 64 KB: x_t tile (f16)
    __shared__ __align__(16) _Float16 zs[B_TILE * H_DIM];  // 32 KB: z_{t-1} spikes (f16)

    const int tid    = threadIdx.x;
    const int wave   = tid >> 5;
    const int lane   = tid & 31;
    const int row    = lane & 15;   // M (A) or N (B/C) index within a tile
    const int hi     = lane >> 4;   // lane half-group
    const int b_base = blockIdx.x * B_TILE;

    const _Float16* w_in16  = ws;
    const _Float16* w_rec16 = ws + WIN_HALVES;
    const _Float16* ow16    = ws + WIN_HALVES + WREC_HALVES;

    // LIF state in WMMA C-layout registers: element (M = r + hi*8, N = row)
    v8f i_acc[NT];
    v8f v_st[NT];
#pragma unroll
    for (int j = 0; j < NT; ++j)
#pragma unroll
        for (int r = 0; r < 8; ++r) { i_acc[j][r] = 0.0f; v_st[j][r] = 0.0f; }

    unsigned int zb = 0u;   // packed spike bits of z_{t-1} (4 tiles x 8 rows)

    for (int t = 0; t < T_STEPS; ++t) {
        __syncthreads();   // previous step's GEMM reads of xs/zs complete

        // ---- publish z_{t-1} (needed by this step's recurrent GEMM) ----
        if (t > 0 && t < T_STEPS - 1) {
#pragma unroll
            for (int j = 0; j < NT; ++j) {
                const int n = (wave * NT + j) * 16 + row;
#pragma unroll
                for (int r = 0; r < 8; ++r)
                    zs[(r + hi * 8) * H_DIM + n] =
                        ((zb >> (j * 8 + r)) & 1u) ? (_Float16)1.0f : (_Float16)0.0f;
            }
        }

        // ---- stage x_t tile: wave w converts batch row w (fp32 -> f16) ----
        if (t < T_STEPS - 1) {
            const float* xr = x + ((size_t)t * B_TOT + b_base + wave) * I_DIM;
#pragma unroll
            for (int c = 0; c < I_DIM / 128; ++c) {     // 16 chunks of 128 cols
                const int col = (c * 32 + lane) * 4;
                f4 f = *(const f4*)(xr + col);
                h4 h;
                h[0] = (_Float16)f[0]; h[1] = (_Float16)f[1];
                h[2] = (_Float16)f[2]; h[3] = (_Float16)f[3];
                *(h4*)(&xs[wave * I_DIM + col]) = h;
            }
        }

        __syncthreads();   // xs / zs ready

        // ---- prefetch next step's x tile into cache (overlaps the GEMMs;
        //      global_prefetch_b8: no counters, dropped on fault) ----
        if (t + 1 < T_STEPS - 1) {
            const char* nx = (const char*)(x + ((size_t)(t + 1) * B_TOT + b_base) * I_DIM);
#pragma unroll
            for (int q = 0; q < 2; ++q)   // 1024 x 128B lines, 512 threads
                __builtin_prefetch(nx + ((size_t)tid * 2 + q) * 128, 0, 3);
        }

        // ---- elementwise LIF on carry (v,i) -> z_t bits, v_new, i decay ----
        zb = 0u;
#pragma unroll
        for (int j = 0; j < NT; ++j) {
#pragma unroll
            for (int r = 0; r < 8; ++r) {
                float vd = v_st[j][r] + 0.1f * (i_acc[j][r] - v_st[j][r]); // DT*TAU_MEM_INV
                bool  sp = vd > 1.0f;                                      // V_TH
                v_st[j][r] = sp ? 0.0f : vd;                               // V_RESET=0
                zb |= sp ? (1u << (j * 8 + r)) : 0u;
            }
        }

        if (t < T_STEPS - 1) {   // last step's i_new is dead
#pragma unroll
            for (int j = 0; j < NT; ++j)
#pragma unroll
                for (int r = 0; r < 8; ++r) i_acc[j][r] *= 0.8f;  // 1 - DT*TAU_SYN_INV

            // ---- i += x_t @ w_in.T  (K = I_DIM), A from LDS ----
            const _Float16* xrow = xs + row * I_DIM + hi * 16;
            for (int kb = 0; kb < I_DIM / 32; ++kb) {
                v16h a = load_a_h(xrow + kb * 32);
#pragma unroll
                for (int j = 0; j < NT; ++j) {
                    const int n_base = (wave * NT + j) * 16;
                    v16h b = *(const v16h*)(w_in16 + (size_t)(n_base + row) * I_DIM
                                            + kb * 32 + hi * 16);
                    i_acc[j] = __builtin_amdgcn_wmma_f32_16x16x32_f16(
                        false, a, false, b, (short)0, i_acc[j], false, false);
                }
            }

            // ---- i += z_{t-1} @ w_rec.T  (K = H_DIM); z==0 at t=0 ----
            if (t > 0) {
                const _Float16* zrow = zs + row * H_DIM + hi * 16;
                for (int kb = 0; kb < H_DIM / 32; ++kb) {
                    v16h a = load_a_h(zrow + kb * 32);
#pragma unroll
                    for (int j = 0; j < NT; ++j) {
                        const int n_base = (wave * NT + j) * 16;
                        v16h b = *(const v16h*)(w_rec16 + (size_t)(n_base + row) * H_DIM
                                                + kb * 32 + hi * 16);
                        i_acc[j] = __builtin_amdgcn_wmma_f32_16x16x32_f16(
                            false, a, false, b, (short)0, i_acc[j], false, false);
                    }
                }
            }
        }
    }

    // ---- publish z_{T-1}, then out = z_T @ out_w.T + out_b ----
#pragma unroll
    for (int j = 0; j < NT; ++j) {
        const int n = (wave * NT + j) * 16 + row;
#pragma unroll
        for (int r = 0; r < 8; ++r)
            zs[(r + hi * 8) * H_DIM + n] =
                ((zb >> (j * 8 + r)) & 1u) ? (_Float16)1.0f : (_Float16)0.0f;
    }
    __syncthreads();

    if (wave < 8) {          // O = 128 -> 8 column tiles
        const int n_base = wave * 16;
        const float bv = out_b[n_base + row];
        v8f acc;
#pragma unroll
        for (int r = 0; r < 8; ++r) acc[r] = bv;

        const _Float16* zrow = zs + row * H_DIM + hi * 16;
        for (int kb = 0; kb < H_DIM / 32; ++kb) {
            v16h a = load_a_h(zrow + kb * 32);
            v16h b = *(const v16h*)(ow16 + (size_t)(n_base + row) * H_DIM
                                    + kb * 32 + hi * 16);
            acc = __builtin_amdgcn_wmma_f32_16x16x32_f16(
                false, a, false, b, (short)0, acc, false, false);
        }
#pragma unroll
        for (int r = 0; r < 8; ++r)
            out[(size_t)(b_base + r + hi * 8) * O_DIM + n_base + row] = acc[r];
    }
}

extern "C" void kernel_launch(void* const* d_in, const int* in_sizes, int n_in,
                              void* d_out, int out_size, void* d_ws, size_t ws_size,
                              hipStream_t stream) {
    const float* x     = (const float*)d_in[0];
    const float* w_in  = (const float*)d_in[1];
    const float* w_rec = (const float*)d_in[2];
    const float* out_w = (const float*)d_in[3];
    const float* out_b = (const float*)d_in[4];
    _Float16* ws  = (_Float16*)d_ws;   // 6.5 MB of f16 weights
    float*    out = (float*)d_out;

    const int total = WIN_HALVES + WREC_HALVES + OW_HALVES;
    convert_weights<<<(total + 255) / 256, 256, 0, stream>>>(w_in, w_rec, out_w, ws);
    lif_kernel<<<B_TOT / B_TILE, WAVES * 32, 0, stream>>>(x, out_b, ws, out);
}
